// HomNeuLIFFSNN_59399397704151
// MI455X (gfx1250) — matrix operand order repaired and checked by reading, not compiled, not run on hardware
//
#include <hip/hip_runtime.h>
#include <hip/hip_bf16.h>

typedef __bf16 v16bf __attribute__((ext_vector_type(16)));
typedef float  v8f   __attribute__((ext_vector_type(8)));

#define B_    128
#define T_    250
#define NIN   700
#define NH    1024
#define NOUT  20
#define KS    22                 // 22 k-steps of 32 (K padded 700 -> 704)
#define RHO_F 0.98019867f        // exp(-0.001/0.05)
#define YSZ   (B_*T_*NOUT)       // 640000 floats (y_out), z_hidden follows

#define XB_ELEMS (T_*8*KS*512)   // 22,528,000 bf16 (packed A fragments)
#define WB_ELEMS (64*KS*512)     //    720,896 bf16 (packed B fragments)

__device__ __forceinline__ unsigned short f2bf(float f) {
  unsigned int u = __float_as_uint(f);
  u += 0x7FFFu + ((u >> 16) & 1u);      // round-to-nearest-even
  return (unsigned short)(u >> 16);
}

// Async copy 16B from global to LDS (gfx1250 GLOBAL_LOAD_ASYNC_TO_LDS_B128, GV mode).
// lds_addr: LDS byte address (low 32 bits of the generic pointer to __shared__ memory,
// per the flat->LDS aperture mapping).
__device__ __forceinline__ void async_ld_b128(unsigned lds_addr, const void* gaddr) {
  asm volatile("global_load_async_to_lds_b128 %0, %1, off"
               :: "v"(lds_addr), "v"((unsigned long long)(size_t)gaddr)
               : "memory");
}
__device__ __forceinline__ void wait_async0() {
  asm volatile("s_wait_asynccnt 0x0" ::: "memory");
}

// ---- pack X [128,250,700] f32 -> A-fragment bf16 layout [t][mt][ks][lane][e] ----
// A 16x32 bf16 fragment: lane<16 holds M=lane, K = {0..7, 16..23};
//                        lane>=16 holds M=lane-16, K = {8..15, 24..31}
__global__ void pack_x_kernel(const float* __restrict__ X,
                              unsigned short* __restrict__ Xb) {
  int idx  = blockIdx.x * 256 + threadIdx.x;
  int e    = idx & 15;
  int lane = (idx >> 4) & 31;
  int rest = idx >> 9;
  int ks   = rest % KS;
  int tm   = rest / KS;
  int mt   = tm & 7;
  int t    = tm >> 3;
  int b    = mt * 16 + (lane & 15);
  int k    = ks * 32 + e + (e >= 8 ? 8 : 0) + (lane >= 16 ? 8 : 0);
  float v  = (k < NIN) ? X[((size_t)b * T_ + t) * NIN + k] : 0.0f;
  Xb[idx]  = f2bf(v);
}

// ---- pack W_in [1024,700] f32 -> B-fragment bf16 layout [nt][ks][lane][e] ----
// B 32x16 bf16 fragment: lane<16 holds N=lane, K = {0..7, 16..23}; lane>=16 holds N=lane-16, K = {8..15, 24..31}
__global__ void pack_w_kernel(const float* __restrict__ W,
                              unsigned short* __restrict__ Wb) {
  int idx  = blockIdx.x * 256 + threadIdx.x;
  int e    = idx & 15;
  int lane = (idx >> 4) & 31;
  int rest = idx >> 9;
  int ks   = rest % KS;
  int nt   = rest / KS;
  int n    = nt * 16 + (lane & 15);
  int k    = ks * 32 + e + (e >= 8 ? 8 : 0) + (lane >= 16 ? 8 : 0);
  float v  = (k < NIN) ? W[(size_t)n * NIN + k] : 0.0f;
  Wb[idx]  = f2bf(v);
}

// ---- fused GEMM + LIF scan: v = rho*v + x_t @ W_in^T ; z = (v>=1); v -= z ----
// grid (8 m-tiles, 16 n-groups), 128 threads = 4 waves; each wave owns one
// 16(batch) x 16(hidden) membrane-state tile in a v8f accumulator for all 250 steps.
// A tiles are double-buffered in LDS via async global->LDS copies (ASYNCcnt).
__global__ void lif_kernel(const unsigned short* __restrict__ Xb,
                           const unsigned short* __restrict__ Wb,
                           float* __restrict__ out) {
  __shared__ __align__(32) unsigned short lds_x[2][KS * 512];   // 2 x 22.5 KB
  const int mt   = blockIdx.x;
  const int ng   = blockIdx.y;
  const int tid  = threadIdx.x;
  const int wave = tid >> 5;
  const int lane = tid & 31;
  float* __restrict__ zout = out + YSZ;

  const unsigned ldsb0 = (unsigned)(size_t)(&lds_x[0][0]);
  const unsigned ldsb1 = (unsigned)(size_t)(&lds_x[1][0]);

  // stage t=0 A fragments into LDS (async)
  {
    const uint4* src = (const uint4*)(Xb + (size_t)mt * (KS * 512));
    for (int i = tid; i < (KS * 512 / 8); i += 128)
      async_ld_b128(ldsb0 + (unsigned)i * 16u, src + i);
    wait_async0();
  }
  __syncthreads();

  const int nt  = ng * 4 + wave;
  const int M0  = (lane >> 4) * 8;          // C/D layout: lanes>=16 hold M=8..15
  const int bb0 = mt * 16 + M0;
  const int n0  = nt * 16 + (lane & 15);
  v8f vst = {};                             // membrane potential tile (persistent)

  for (int t = 0; t < T_; ++t) {
    // async-stage t+1 while computing t (one barrier per iteration)
    if (t + 1 < T_) {
      const uint4* src = (const uint4*)(Xb + ((size_t)(t + 1) * 8 + mt) * (KS * 512));
      const unsigned dbase = ((t + 1) & 1) ? ldsb1 : ldsb0;
      for (int i = tid; i < (KS * 512 / 8); i += 128)
        async_ld_b128(dbase + (unsigned)i * 16u, src + i);
      if (t + 2 < T_)
        __builtin_prefetch(Xb + ((size_t)(t + 2) * 8 + mt) * (KS * 512) + tid * 176, 0, 0);
    }

    v8f acc = vst * RHO_F;                  // leak folded into the accumulator init
    const unsigned short* xb = lds_x[t & 1];
#pragma unroll 2
    for (int ks = 0; ks < KS; ++ks) {
      v16bf a = *(const v16bf*)(xb + ks * 512 + lane * 16);                       // ds_load
      v16bf b = *(const v16bf*)(Wb + ((size_t)nt * KS + ks) * 512 + lane * 16);   // L2-hot
      acc = __builtin_amdgcn_wmma_f32_16x16x32_bf16(
                false, a, false, b, (short)0, acc, false, false);
    }

    // spike + soft reset + store z (fp32, fully covers z_hidden)
#pragma unroll
    for (int r = 0; r < 8; ++r) {
      float z = (acc[r] >= 1.0f) ? 1.0f : 0.0f;
      acc[r] -= z;
      zout[((size_t)(bb0 + r) * T_ + t) * NH + n0] = z;
    }
    vst = acc;

    wait_async0();                          // t+1 buffer resident before next reads
    __syncthreads();
  }
}

// ---- U[bt][o] = sum_h z[bt][h] * w_out[o][h]  (z row is wave-uniform) ----
__global__ void ugemm_kernel(const float* __restrict__ out,
                             const float* __restrict__ w_out,
                             float* __restrict__ U) {
  int tid = blockIdx.x * 256 + threadIdx.x;
  int o   = tid & 31;
  int bt  = tid >> 5;
  if (bt >= B_ * T_ || o >= NOUT) return;
  const float* z = out + YSZ + (size_t)bt * NH;
  const float* w = w_out + (size_t)o * NH;
  float s = 0.f;
  for (int h = 0; h < NH; h += 4)
    s += z[h] * w[h] + z[h+1] * w[h+1] + z[h+2] * w[h+2] + z[h+3] * w[h+3];
  U[(size_t)bt * NOUT + o] = s;
}

// ---- output leaky integrator scan: vout = rho*vout + U[t] ----
__global__ void vscan_kernel(const float* __restrict__ U,
                             float* __restrict__ out) {
  int tid = blockIdx.x * 256 + threadIdx.x;
  if (tid >= B_ * NOUT) return;
  int b = tid / NOUT, o = tid % NOUT;
  float v = 0.f;
  for (int t = 0; t < T_; ++t) {
    v = RHO_F * v + U[((size_t)b * T_ + t) * NOUT + o];
    out[((size_t)b * T_ + t) * NOUT + o] = v;
  }
}

extern "C" void kernel_launch(void* const* d_in, const int* in_sizes, int n_in,
                              void* d_out, int out_size, void* d_ws, size_t ws_size,
                              hipStream_t stream) {
  const float* X     = (const float*)d_in[0];
  const float* w_in  = (const float*)d_in[1];
  const float* w_out = (const float*)d_in[2];
  float* out = (float*)d_out;

  unsigned short* Xb = (unsigned short*)d_ws;            // 45.06 MB bf16
  unsigned short* Wb = Xb + XB_ELEMS;                    //  1.44 MB bf16
  float* U           = (float*)(Wb + WB_ELEMS);          //  2.56 MB f32

  pack_w_kernel<<<WB_ELEMS / 256, 256, 0, stream>>>(w_in, Wb);
  pack_x_kernel<<<XB_ELEMS / 256, 256, 0, stream>>>(X, Xb);
  lif_kernel<<<dim3(8, 16), 128, 0, stream>>>(Xb, Wb, out);
  ugemm_kernel<<<(B_ * T_ * 32) / 256, 256, 0, stream>>>(out, w_out, U);
  vscan_kernel<<<(B_ * NOUT + 255) / 256, 256, 0, stream>>>(U, out);
}
